// MiniTrajectoryPredictor_22127671509836
// MI455X (gfx1250) — matrix operand order repaired and checked by reading, not compiled, not run on hardware
//
#include <hip/hip_runtime.h>

// ---------------------------------------------------------------------------
// MiniTrajectoryPredictor on gfx1250 (MI455X, wave32, WMMA bf16)
// 4096 independent GRU sequences; 16 rows per workgroup, h resident in LDS
// for all 72 recurrent steps. 512 threads = 16 waves; each wave owns 32
// h-columns (2 N-tiles) and the matching r/z/n gate columns, so gates and the
// h update are wave-local. v_wmma_f32_16x16x32_bf16 with f32 accumulate.
// B fragments address off 2 base pointers using compile-time immediate
// offsets (gate blocks / N-tiles fit in the 24-bit signed IOFFSET).
// ---------------------------------------------------------------------------

typedef __attribute__((ext_vector_type(16))) __bf16 v16bf;
typedef __attribute__((ext_vector_type(8)))  float  v8f;

#define NB   32
#define NT   24
#define NO   128
#define NH   512
#define NCTX 256
#define NHOR 48
#define GATE (3 * NH)           // 1536
#define GATE_ELEMS (GATE * NH)  // 786432
#define HH_ELEMS   (NH * NH)    // 262144

static __device__ __forceinline__ v8f wmma_bf16(v16bf a, v16bf b, v8f c) {
  return __builtin_amdgcn_wmma_f32_16x16x32_bf16(
      /*neg_a=*/false, a, /*neg_b=*/false, b,
      /*c_mod=*/(short)0, c, /*reuse_a=*/false, /*reuse_b=*/false);
}

// A fragment (16x32 bf16). Caller passes &buf[row][lhalf*16]; the two 16-byte
// chunks sit at +kb and +kb+8 (contiguous K, matching B).
static __device__ __forceinline__ v16bf load_a(const __bf16* base, int kb) {
  union { v16bf v; uint4 q[2]; } u;
  u.q[0] = *(const uint4*)(base + kb);
  u.q[1] = *(const uint4*)(base + kb + 8);
  return u.v;
}

// B fragment: base = W + n*NH + lhalf*16; `off` is a (mostly compile-time)
// element offset that the backend folds into the load's 24-bit IOFFSET.
static __device__ __forceinline__ v16bf load_b(const __bf16* base, int off) {
  return *(const v16bf*)(base + off);
}

static __device__ __forceinline__ float sigmoidf(float x) {
  return 1.0f / (1.0f + __expf(-x));
}

__global__ void cvt_bf16_kernel(const float* __restrict__ s, __bf16* __restrict__ d, int n) {
  int i = blockIdx.x * blockDim.x + threadIdx.x;
  if (i < n) d[i] = (__bf16)s[i];
}

__global__ __launch_bounds__(512, 1)
void traj_gru_kernel(const float* __restrict__ z_ctx,
                     const float* __restrict__ traj,
                     const float* __restrict__ W_emb, const float* __restrict__ b_emb,
                     const float* __restrict__ W_ctx, const float* __restrict__ b_ctx,
                     const float* __restrict__ bi_enc, const float* __restrict__ bh_enc,
                     const float* __restrict__ bi_cell, const float* __restrict__ bh_cell,
                     const float* __restrict__ b1,
                     const float* __restrict__ W2, const float* __restrict__ b2,
                     const __bf16* __restrict__ WiE, const __bf16* __restrict__ WhE,
                     const __bf16* __restrict__ WiC, const __bf16* __restrict__ WhC,
                     const __bf16* __restrict__ W1b,
                     float* __restrict__ out) {
  __shared__ float  h_s[16][NH];    // 32 KB  f32 hidden state
  __shared__ __bf16 hb_s[16][NH];   // 16 KB  bf16 mirror of h (A operand)
  __shared__ __bf16 x_s[16][NH];    // 16 KB  step input / gelu-hidden
  __shared__ float  ctx_s[NH];      //  2 KB
  __shared__ float  st_s[16][2];
  __shared__ float  red_s[32][16];  //  2 KB  W2 head reduction

  const int tid   = threadIdx.x;
  const int wave  = tid >> 5;       // 0..15
  const int lane  = tid & 31;
  const int lidx  = lane & 15;      // A row / B,D column
  const int lhalf = lane >> 4;      // K-half select
  const int cb    = wave * 32;      // this wave's h-column base (2 N-tiles)

  const int rowbase = blockIdx.x * 16;
  const int b_      = rowbase >> 7;
  const int o_base  = rowbase & 127;

  // Per-wave A fragment base pointers (row = lidx, K-half folded in).
  const __bf16* aXbase = &x_s[lidx][lhalf * 16];
  const __bf16* aHbase = &hb_s[lidx][lhalf * 16];

  // Compile-time element offsets for B fragments (fold into IOFFSET):
  constexpr int T1 = 16 * NH;      // next N-tile        (16 KB bytes)
  constexpr int G1 = NH * NH;      // z-gate block       (512 KB bytes)
  constexpr int G2 = 2 * NH * NH;  // n-gate block       (1 MB bytes)

  // ---- ctx = z_ctx[b] @ W_ctx^T + b_ctx ----
  for (int j = tid; j < NH; j += 512) {
    float acc = b_ctx[j];
    const float* zr = z_ctx + (size_t)b_ * NCTX;
    const float* wr = W_ctx + (size_t)j * NCTX;
    for (int k = 0; k < NCTX; ++k) acc += zr[k] * wr[k];
    ctx_s[j] = acc;
  }
  // ---- h0 = 0 (both copies) ----
  for (int i = tid; i < 16 * NH; i += 512) {
    ((float*)h_s)[i]   = 0.0f;
    ((__bf16*)hb_s)[i] = (__bf16)0.0f;
  }
  __syncthreads();

  // ---- one GRU step over this wave's 32 columns ----
  auto gru_step = [&](const __bf16* Wi, const __bf16* Wh,
                      const float* bi, const float* bh) {
    const __bf16* wiB = Wi + (size_t)(cb + lidx) * NH + lhalf * 16;
    const __bf16* whB = Wh + (size_t)(cb + lidx) * NH + lhalf * 16;
    v8f accR[2], accZ[2], accNi[2], accNh[2];
#pragma unroll
    for (int j = 0; j < 2; ++j) {
      accR[j]  = (v8f){0,0,0,0,0,0,0,0};
      accZ[j]  = (v8f){0,0,0,0,0,0,0,0};
      accNi[j] = (v8f){0,0,0,0,0,0,0,0};
      accNh[j] = (v8f){0,0,0,0,0,0,0,0};
    }
#pragma unroll 1
    for (int kb = 0; kb < NH; kb += 32) {
      const v16bf ax = load_a(aXbase, kb);
      const v16bf ah = load_a(aHbase, kb);
      // Issue all 12 B-fragment loads before consuming any (one clause).
      const v16bf bir0 = load_b(wiB, kb);
      const v16bf bir1 = load_b(wiB, kb + T1);
      const v16bf biz0 = load_b(wiB, kb + G1);
      const v16bf biz1 = load_b(wiB, kb + G1 + T1);
      const v16bf bin0 = load_b(wiB, kb + G2);
      const v16bf bin1 = load_b(wiB, kb + G2 + T1);
      const v16bf bhr0 = load_b(whB, kb);
      const v16bf bhr1 = load_b(whB, kb + T1);
      const v16bf bhz0 = load_b(whB, kb + G1);
      const v16bf bhz1 = load_b(whB, kb + G1 + T1);
      const v16bf bhn0 = load_b(whB, kb + G2);
      const v16bf bhn1 = load_b(whB, kb + G2 + T1);
      // ax-group then ah-group (A operand reuse between consecutive wmmas).
      accR[0]  = wmma_bf16(ax, bir0, accR[0]);
      accR[1]  = wmma_bf16(ax, bir1, accR[1]);
      accZ[0]  = wmma_bf16(ax, biz0, accZ[0]);
      accZ[1]  = wmma_bf16(ax, biz1, accZ[1]);
      accNi[0] = wmma_bf16(ax, bin0, accNi[0]);
      accNi[1] = wmma_bf16(ax, bin1, accNi[1]);
      accR[0]  = wmma_bf16(ah, bhr0, accR[0]);
      accR[1]  = wmma_bf16(ah, bhr1, accR[1]);
      accZ[0]  = wmma_bf16(ah, bhz0, accZ[0]);
      accZ[1]  = wmma_bf16(ah, bhz1, accZ[1]);
      accNh[0] = wmma_bf16(ah, bhn0, accNh[0]);
      accNh[1] = wmma_bf16(ah, bhn1, accNh[1]);
    }
    __syncthreads();   // all waves done reading h/x before h is rewritten
#pragma unroll
    for (int j = 0; j < 2; ++j) {
      const int col = cb + j * 16 + lidx;
      const float biR = bi[col],        bhR = bh[col];
      const float biZ = bi[NH + col],   bhZ = bh[NH + col];
      const float biN = bi[2*NH + col], bhN = bh[2*NH + col];
#pragma unroll
      for (int v = 0; v < 8; ++v) {
        const int m = v + 8 * lhalf;
        const float r = sigmoidf(accR[j][v] + biR + bhR);
        const float z = sigmoidf(accZ[j][v] + biZ + bhZ);
        const float n = tanhf((accNi[j][v] + biN) + r * (accNh[j][v] + bhN));
        const float hn = (1.0f - z) * n + z * h_s[m][col];
        h_s[m][col]  = hn;
        hb_s[m][col] = (__bf16)hn;
      }
    }
    __syncthreads();
  };

  // =============================== encoder ================================
  for (int t = 0; t < NT; ++t) {
    for (int i = tid; i < 16 * NH; i += 512) {
      const int m = i >> 9, j = i & (NH - 1);
      const size_t base = (((size_t)b_ * NT + t) * NO + (o_base + m)) * 2;
      const float s0 = traj[base + 0], s1 = traj[base + 1];
      x_s[m][j] = (__bf16)(W_emb[j * 2 + 0] * s0 + W_emb[j * 2 + 1] * s1 + b_emb[j]);
    }
    __syncthreads();
    gru_step(WiE, WhE, bi_enc, bh_enc);
  }

  // h += ctx ; state0 = seq[:, T-1]
  for (int i = tid; i < 16 * NH; i += 512) {
    const int m = i >> 9, j = i & (NH - 1);
    const float hn = h_s[m][j] + ctx_s[j];
    h_s[m][j]  = hn;
    hb_s[m][j] = (__bf16)hn;
  }
  if (tid < 32) {
    const int m = tid >> 1, c = tid & 1;
    st_s[m][c] = traj[(((size_t)b_ * NT + (NT - 1)) * NO + (o_base + m)) * 2 + c];
  }
  __syncthreads();

  // =============================== decoder ================================
  for (int t = 0; t < NHOR; ++t) {
    // inp = state @ W_emb^T + b_emb
    for (int i = tid; i < 16 * NH; i += 512) {
      const int m = i >> 9, j = i & (NH - 1);
      const float s0 = st_s[m][0], s1 = st_s[m][1];
      x_s[m][j] = (__bf16)(W_emb[j * 2 + 0] * s0 + W_emb[j * 2 + 1] * s1 + b_emb[j]);
    }
    __syncthreads();
    gru_step(WiC, WhC, bi_cell, bh_cell);

    // hid = gelu(h @ W1^T + b1), wave-local 32 columns, exact erf gelu
    {
      const __bf16* w1B = W1b + (size_t)(cb + lidx) * NH + lhalf * 16;
      v8f accH0 = (v8f){0,0,0,0,0,0,0,0};
      v8f accH1 = (v8f){0,0,0,0,0,0,0,0};
#pragma unroll 1
      for (int kb = 0; kb < NH; kb += 32) {
        const v16bf ah = load_a(aHbase, kb);
        const v16bf b0 = load_b(w1B, kb);
        const v16bf b1f = load_b(w1B, kb + T1);
        accH0 = wmma_bf16(ah, b0,  accH0);
        accH1 = wmma_bf16(ah, b1f, accH1);
      }
      __syncthreads();   // gi GEMM readers of x_s are done; safe to overwrite
      v8f accH[2] = {accH0, accH1};
#pragma unroll
      for (int j = 0; j < 2; ++j) {
        const int col = cb + j * 16 + lidx;
        const float bb = b1[col];
#pragma unroll
        for (int v = 0; v < 8; ++v) {
          const int m = v + 8 * lhalf;
          const float x = accH[j][v] + bb;
          x_s[m][col] = (__bf16)(0.5f * x * (1.0f + erff(x * 0.70710678118654752f)));
        }
      }
    }
    __syncthreads();

    // d = hid @ W2^T + b2 : 32 dots of length 512, 16 threads per dot
    {
      const int m = tid >> 5, c = (tid >> 4) & 1, part = tid & 15;
      const float* w2r = W2 + (size_t)c * NH;
      float acc = 0.0f;
      const int k0 = part * 32;
      for (int k = k0; k < k0 + 32; ++k) acc += (float)x_s[m][k] * w2r[k];
      red_s[m * 2 + c][part] = acc;
    }
    __syncthreads();
    if (tid < 32) {
      const int m = tid >> 1, c = tid & 1;
      float acc = b2[c];
#pragma unroll
      for (int p = 0; p < 16; ++p) acc += red_s[tid][p];
      const float d = 2.0f * tanhf(acc * 0.5f);   // MAX_STEP = 2
      float s = st_s[m][c] + d;
      if (c == 0) s = fminf(90.0f, fmaxf(-90.0f, s));
      else        s = s - floorf(s * (1.0f / 360.0f)) * 360.0f;   // python mod
      st_s[m][c] = s;
      out[((((size_t)b_ * NHOR + t) * NO) + (o_base + m)) * 2 + c] = s;
    }
    __syncthreads();
  }
}

extern "C" void kernel_launch(void* const* d_in, const int* in_sizes, int n_in,
                              void* d_out, int out_size, void* d_ws, size_t ws_size,
                              hipStream_t stream) {
  const float* z_ctx   = (const float*)d_in[0];
  const float* traj    = (const float*)d_in[1];
  const float* W_emb   = (const float*)d_in[2];
  const float* b_emb   = (const float*)d_in[3];
  const float* W_ctx   = (const float*)d_in[4];
  const float* b_ctx   = (const float*)d_in[5];
  const float* Wi_enc  = (const float*)d_in[6];
  const float* Wh_enc  = (const float*)d_in[7];
  const float* bi_enc  = (const float*)d_in[8];
  const float* bh_enc  = (const float*)d_in[9];
  const float* Wi_cell = (const float*)d_in[10];
  const float* Wh_cell = (const float*)d_in[11];
  const float* bi_cell = (const float*)d_in[12];
  const float* bh_cell = (const float*)d_in[13];
  const float* W1      = (const float*)d_in[14];
  const float* b1      = (const float*)d_in[15];
  const float* W2      = (const float*)d_in[16];
  const float* b2      = (const float*)d_in[17];
  float* out = (float*)d_out;

  // bf16 weight mirrors in workspace (~6.8 MB)
  __bf16* ws   = (__bf16*)d_ws;
  __bf16* wiE  = ws;
  __bf16* whE  = ws + 1 * (size_t)GATE_ELEMS;
  __bf16* wiC  = ws + 2 * (size_t)GATE_ELEMS;
  __bf16* whC  = ws + 3 * (size_t)GATE_ELEMS;
  __bf16* w1b  = ws + 4 * (size_t)GATE_ELEMS;

  const int thr = 256;
  cvt_bf16_kernel<<<(GATE_ELEMS + thr - 1) / thr, thr, 0, stream>>>(Wi_enc,  wiE, GATE_ELEMS);
  cvt_bf16_kernel<<<(GATE_ELEMS + thr - 1) / thr, thr, 0, stream>>>(Wh_enc,  whE, GATE_ELEMS);
  cvt_bf16_kernel<<<(GATE_ELEMS + thr - 1) / thr, thr, 0, stream>>>(Wi_cell, wiC, GATE_ELEMS);
  cvt_bf16_kernel<<<(GATE_ELEMS + thr - 1) / thr, thr, 0, stream>>>(Wh_cell, whC, GATE_ELEMS);
  cvt_bf16_kernel<<<(HH_ELEMS   + thr - 1) / thr, thr, 0, stream>>>(W1,      w1b, HH_ELEMS);

  // 4096 rows / 16 rows-per-WG = 256 workgroups, 16 waves each
  traj_gru_kernel<<<256, 512, 0, stream>>>(
      z_ctx, traj, W_emb, b_emb, W_ctx, b_ctx,
      bi_enc, bh_enc, bi_cell, bh_cell, b1, W2, b2,
      wiE, whE, wiC, whC, w1b, out);
}
